// GroupedQueryAttention_62182536512212
// MI455X (gfx1250) — compile-verified, matrix operand order checked by reading, not run on hardware
//
#include <hip/hip_runtime.h>
#include <hip/hip_bf16.h>
#include <stdint.h>

// ---------------------------------------------------------------------------
// GQA forward for MI455X (gfx1250, wave32, WMMA bf16 16x16x32, f32 accum).
//   B=2, S=2048, F=4096, HQ=32, HKV=8, D=128, REP=4
// Pipeline:
//   1) Q = query @ wq * (1/sqrt(D)) ; K = value @ wk ; V = value @ wv  (bf16 ws)
//   2) flash attention (online softmax, async K-tile loads to LDS), bf16 ws
//   3) out = attn @ wo (f32)
// Workspace: Q 32MB + K 8MB + V 8MB + attn 32MB = 80MB (assumed < ws_size).
// ---------------------------------------------------------------------------

#define BB   2
#define SS   2048
#define FF   4096
#define HQ   32
#define HKV  8
#define DD   128
#define QKD  (HQ * DD)    // 4096
#define KVD  (HKV * DD)   // 1024

typedef __attribute__((ext_vector_type(16))) __bf16    v16bf;
typedef __attribute__((ext_vector_type(8)))  float     v8f;
typedef __attribute__((ext_vector_type(8)))  unsigned  v8u;

static_assert(sizeof(v16bf) == 32, "v16bf must be 32B");

__device__ __forceinline__ unsigned short f2bf(float f) {
    unsigned u = __float_as_uint(f);
    unsigned r = u + 0x7FFFu + ((u >> 16) & 1u);   // round-to-nearest-even
    return (unsigned short)(r >> 16);
}
__device__ __forceinline__ unsigned pack2bf(float lo, float hi) {
    return (unsigned)f2bf(lo) | ((unsigned)f2bf(hi) << 16);
}
__device__ __forceinline__ v8f vzero8() {
    v8f z = {0.f,0.f,0.f,0.f,0.f,0.f,0.f,0.f};
    return z;
}

// Load a 16x32 bf16 WMMA A/B fragment from a row-major [rows x stride] bf16
// array (LDS or global). Per the CDNA5 16-bit A layout, lane (m = lane%16,
// h = lane/16) holds two contiguous 8-element runs: k = kOff+8h.. and
// k = kOff+16+8h.. -> exactly two 16-byte loads per lane.
// (B fragments use the same pattern when the B matrix is stored transposed,
//  i.e. row index = output column n.)
__device__ __forceinline__ v16bf ldfrag(const unsigned short* base, int row,
                                        int stride, int kOff) {
    int lane = threadIdx.x & 31;
    int m  = lane & 15;
    int hh = lane >> 4;
    const unsigned short* p = base + (size_t)(row + m) * stride + kOff + 8 * hh;
    uint4 lo = *(const uint4*)p;          // k = kOff+8h   .. +7
    uint4 hi = *(const uint4*)(p + 16);   // k = kOff+16+8h.. +7
    v8u u = { lo.x, lo.y, lo.z, lo.w, hi.x, hi.y, hi.z, hi.w };
    return __builtin_bit_cast(v16bf, u);
}

__device__ __forceinline__ v8f wmma_bf16(v16bf a, v16bf b, v8f c) {
    return __builtin_amdgcn_wmma_f32_16x16x32_bf16(
        /*neg_a=*/false, a, /*neg_b=*/false, b,
        /*c_mod=*/(short)0, c, /*reuse_a=*/false, /*reuse_b=*/false);
}

// 16-lane-half reductions (wave32: xor masks 1,2,4,8 stay within each half,
// matching the C-fragment layout where a matrix row lives on 16 lanes).
__device__ __forceinline__ float hmax16(float v) {
    for (int m = 1; m <= 8; m <<= 1) v = fmaxf(v, __shfl_xor(v, m, 32));
    return v;
}
__device__ __forceinline__ float hsum16(float v) {
    for (int m = 1; m <= 8; m <<= 1) v += __shfl_xor(v, m, 32);
    return v;
}

// ---------------------------------------------------------------------------
// Generic tiled GEMM:  C[M x N] = A[M x K] * B[K x N]  (B is f32 row-major;
// A is f32 or bf16; C is f32 or bf16, scaled by outScale).
// Workgroup tile 128(M) x 128(N), 8 waves as 2x4, wave tile 64x32, K step 32.
// Per wave per K-step: 6 fragment loads (12 ds_load_b128) feed 8 WMMAs.
// ---------------------------------------------------------------------------
template <bool A_BF16, bool OUT_BF16>
__global__ __launch_bounds__(256)
void gemm_wmma(const void* __restrict__ Aptr, int lda,
               const float* __restrict__ B, int ldb,
               void* __restrict__ Cptr, int ldc,
               int K, float outScale) {
    __shared__ unsigned short sA [128][40];  // [m][k]  pad 40 (80B rows)
    __shared__ unsigned short sBT[128][40];  // [n][k]  transposed B tile

    const int nBase = blockIdx.x * 128;
    const int mBase = blockIdx.y * 128;
    const int w     = threadIdx.x >> 5;      // 8 waves
    const int wm    = w >> 2;                // 0..1  -> 64 rows each
    const int wn    = w & 3;                 // 0..3  -> 32 cols each

    v8f acc[4][2];
#pragma unroll
    for (int i = 0; i < 4; ++i)
#pragma unroll
        for (int j = 0; j < 2; ++j) acc[i][j] = vzero8();

    for (int kb = 0; kb < K; kb += 32) {
        __syncthreads();
        // --- stage A tile [128 x 32] -> bf16 row-major LDS ---
        {
            const int row = threadIdx.x >> 1;         // 0..127
            const int col = (threadIdx.x & 1) * 16;   // 0 or 16
            if (A_BF16) {
                const unsigned short* Ab = (const unsigned short*)Aptr;
                const unsigned short* p = Ab + (size_t)(mBase + row) * lda + kb + col;
                *(uint4*)&sA[row][col]     = *(const uint4*)p;
                *(uint4*)&sA[row][col + 8] = *(const uint4*)(p + 8);
            } else {
                const float* Af = (const float*)Aptr;
                const float* p = Af + (size_t)(mBase + row) * lda + kb + col;
                float4 f0 = *(const float4*)p;
                float4 f1 = *(const float4*)(p + 4);
                float4 f2 = *(const float4*)(p + 8);
                float4 f3 = *(const float4*)(p + 12);
                unsigned* d = (unsigned*)&sA[row][col];
                d[0] = pack2bf(f0.x, f0.y); d[1] = pack2bf(f0.z, f0.w);
                d[2] = pack2bf(f1.x, f1.y); d[3] = pack2bf(f1.z, f1.w);
                d[4] = pack2bf(f2.x, f2.y); d[5] = pack2bf(f2.z, f2.w);
                d[6] = pack2bf(f3.x, f3.y); d[7] = pack2bf(f3.z, f3.w);
            }
        }
        // --- stage B tile [32 x 128] -> bf16 transposed LDS [n][k] ---
        {
            const int k0 = (threadIdx.x >> 4) * 2;    // 0,2,..,30
            const int nb = (threadIdx.x & 15) * 8;    // 0..120
            const float* Br0 = B + (size_t)(kb + k0) * ldb + nBase + nb;
            const float* Br1 = Br0 + ldb;
            float4 a0 = *(const float4*)Br0, a1 = *(const float4*)(Br0 + 4);
            float4 b0 = *(const float4*)Br1, b1 = *(const float4*)(Br1 + 4);
            float r0[8] = {a0.x,a0.y,a0.z,a0.w,a1.x,a1.y,a1.z,a1.w};
            float r1[8] = {b0.x,b0.y,b0.z,b0.w,b1.x,b1.y,b1.z,b1.w};
#pragma unroll
            for (int j = 0; j < 8; ++j)
                *(unsigned*)&sBT[nb + j][k0] = pack2bf(r0[j], r1[j]);
        }
        __syncthreads();

        v16bf af[4], bf[2];
#pragma unroll
        for (int i = 0; i < 4; ++i)
            af[i] = ldfrag(&sA[0][0], wm * 64 + 16 * i, 40, 0);
#pragma unroll
        for (int j = 0; j < 2; ++j)
            bf[j] = ldfrag(&sBT[0][0], wn * 32 + 16 * j, 40, 0);
#pragma unroll
        for (int i = 0; i < 4; ++i)
#pragma unroll
            for (int j = 0; j < 2; ++j)
                acc[i][j] = wmma_bf16(af[i], bf[j], acc[i][j]);
    }

    // --- epilogue: C-frag layout = lane(n=lane%16, h=lane/16), vgpr r -> row r+8h
    const int lane = threadIdx.x & 31;
    const int n    = lane & 15;
    const int hh   = lane >> 4;
#pragma unroll
    for (int i = 0; i < 4; ++i)
#pragma unroll
        for (int j = 0; j < 2; ++j) {
            v8f c = acc[i][j];
            const int gm0 = mBase + wm * 64 + i * 16 + 8 * hh;
            const int gn  = nBase + wn * 32 + j * 16 + n;
#pragma unroll
            for (int r = 0; r < 8; ++r) {
                const float v = c[r] * outScale;
                const size_t idx = (size_t)(gm0 + r) * ldc + gn;
                if (OUT_BF16) ((unsigned short*)Cptr)[idx] = f2bf(v);
                else          ((float*)Cptr)[idx] = v;
            }
        }
}

// ---------------------------------------------------------------------------
// Flash attention: workgroup = 128 q rows of one (b,h); 8 waves x 16 q rows.
// Q lives in registers as 4 A-fragments (reused over all 32 K tiles).
// K tile [64 x 128] is DMA'd into LDS with global_load_async_to_lds_b128
// (ASYNCcnt path, no VGPR round-trip); scores B-frags read it directly since
// B^T = K. V tile is repacked transposed [d][k] so PV B-frags are contiguous.
// ---------------------------------------------------------------------------
__global__ __launch_bounds__(256)
void fa_kernel(const unsigned short* __restrict__ Q,
               const unsigned short* __restrict__ Kb,
               const unsigned short* __restrict__ Vb,
               unsigned short* __restrict__ O) {
    __shared__ unsigned short sK [64][136];    // [k][d]   272B rows
    __shared__ unsigned short sVt[128][72];    // [d][k]   144B rows
    __shared__ unsigned short sP [8][16][72];  // per-wave P scratch [q][k]

    const int qblk = blockIdx.x;       // 0..15
    const int h    = blockIdx.y;       // 0..31
    const int bi   = blockIdx.z;       // 0..1
    const int hk   = h >> 2;           // GQA: REP=4
    const int w    = threadIdx.x >> 5;
    const int lane = threadIdx.x & 31;
    const int nl   = lane & 15;
    const int hh   = lane >> 4;
    const int q0   = qblk * 128 + w * 16;

    // Q fragments straight from global (bf16, already scaled by 1/sqrt(D)).
    const unsigned short* qbase = Q + (size_t)(bi * SS + q0) * QKD + h * DD;
    v16bf qf[4];
#pragma unroll
    for (int j = 0; j < 4; ++j) qf[j] = ldfrag(qbase, 0, QKD, 32 * j);

    float m_i[8], l_i[8];
    v8f   o[8];
#pragma unroll
    for (int r = 0; r < 8; ++r) { m_i[r] = -1e30f; l_i[r] = 0.f; }
#pragma unroll
    for (int ds = 0; ds < 8; ++ds) o[ds] = vzero8();

    // Per-thread async-copy geometry for the K tile (each lane copies 64B).
    const int kr  = threadIdx.x >> 2;            // 0..63  (k row)
    const int kcb = (threadIdx.x & 3) * 32;      // d col base
    const uint64_t kBase = (uint64_t)(const void*)Kb;
    // LDS byte offset of this thread's destination (low 32 bits of the
    // generic pointer to an LDS object are the LDS offset on CDNA5).
    const unsigned kLds = (unsigned)(size_t)(const void*)&sK[kr][kcb];

    for (int kt = 0; kt < SS / 64; ++kt) {
        const int k0 = kt * 64;
        __syncthreads();

        // --- K tile: async DMA global -> LDS (ASYNCcnt), 4 x b128 per lane ---
        {
            const unsigned gOff =
                (unsigned)(((size_t)(bi * SS + k0 + kr) * KVD + hk * DD + kcb) * 2);
#pragma unroll
            for (int i = 0; i < 4; ++i)
                asm volatile(
                    "global_load_async_to_lds_b128 %0, %1, %2 offset:%3"
                    :: "v"(kLds), "v"(gOff), "s"(kBase), "i"(i * 16)
                    : "memory");
        }
        // --- V tile: load + transpose repack into sVt[d][k] ---
        {
            const int r0 = (threadIdx.x >> 3) * 2;    // 0,2,..,62
            const int cb = (threadIdx.x & 7) * 16;    // d 0..112
            const unsigned short* vp =
                Vb + (size_t)(bi * SS + k0 + r0) * KVD + hk * DD + cb;
            unsigned short a[16], c[16];
            *(uint4*)(a)     = *(const uint4*)(vp);
            *(uint4*)(a + 8) = *(const uint4*)(vp + 8);
            *(uint4*)(c)     = *(const uint4*)(vp + KVD);
            *(uint4*)(c + 8) = *(const uint4*)(vp + KVD + 8);
#pragma unroll
            for (int j = 0; j < 16; ++j)
                *(unsigned*)&sVt[cb + j][r0] =
                    (unsigned)a[j] | ((unsigned)c[j] << 16);
        }
        asm volatile("s_wait_asynccnt 0" ::: "memory");
        __syncthreads();

        // prefetch next K tile into the cache hierarchy
        if (kt + 1 < SS / 64) {
            const unsigned short* nk =
                Kb + (size_t)(bi * SS + k0 + 64 + kr) * KVD + hk * DD + kcb;
            __builtin_prefetch(nk, 0, 1);
        }

        // --- scores S = Q * K^T : 16 q x 64 k per wave ---
        v8f sv[4];
#pragma unroll
        for (int ns = 0; ns < 4; ++ns) {
            sv[ns] = vzero8();
#pragma unroll
            for (int j = 0; j < 4; ++j) {
                v16bf kf = ldfrag(&sK[0][0], ns * 16, 136, 32 * j);
                sv[ns] = wmma_bf16(qf[j], kf, sv[ns]);
            }
        }

        // --- online softmax (per q row = vgpr index r, 16 lanes/row) ---
        float sc[8];
#pragma unroll
        for (int r = 0; r < 8; ++r) {
            float mx = sv[0][r];
            mx = fmaxf(mx, sv[1][r]);
            mx = fmaxf(mx, sv[2][r]);
            mx = fmaxf(mx, sv[3][r]);
            mx = hmax16(mx);
            const float mt = fmaxf(m_i[r], mx);
            sc[r] = __expf(m_i[r] - mt);
            float rs = 0.f;
#pragma unroll
            for (int ns = 0; ns < 4; ++ns) {
                float p = __expf(sv[ns][r] - mt);
                sv[ns][r] = p;
                rs += p;
            }
            rs = hsum16(rs);
            l_i[r] = l_i[r] * sc[r] + rs;
            m_i[r] = mt;
        }
#pragma unroll
        for (int ds = 0; ds < 8; ++ds)
#pragma unroll
            for (int r = 0; r < 8; ++r) o[ds][r] *= sc[r];

        // --- P: C-layout -> A-layout via per-wave LDS scratch ---
#pragma unroll
        for (int ns = 0; ns < 4; ++ns)
#pragma unroll
            for (int r = 0; r < 8; ++r)
                sP[w][r + 8 * hh][ns * 16 + nl] = f2bf(sv[ns][r]);
        // same-wave LDS ops are in-order: safe to read back without barrier

        // --- O += P * V ---
#pragma unroll
        for (int kc = 0; kc < 2; ++kc) {
            v16bf pf = ldfrag(&sP[w][0][0], 0, 72, 32 * kc);
#pragma unroll
            for (int ds = 0; ds < 8; ++ds) {
                v16bf vf = ldfrag(&sVt[0][0], ds * 16, 72, 32 * kc);
                o[ds] = wmma_bf16(pf, vf, o[ds]);
            }
        }
    }

    // --- epilogue: normalize by l_i and store bf16 attention output ---
#pragma unroll
    for (int r = 0; r < 8; ++r) {
        const float inv = 1.0f / l_i[r];
        const int q = q0 + r + 8 * hh;
        unsigned short* op = O + (size_t)(bi * SS + q) * QKD + h * DD + nl;
#pragma unroll
        for (int ds = 0; ds < 8; ++ds)
            op[ds * 16] = f2bf(o[ds][r] * inv);
    }
}

// ---------------------------------------------------------------------------
extern "C" void kernel_launch(void* const* d_in, const int* in_sizes, int n_in,
                              void* d_out, int out_size, void* d_ws, size_t ws_size,
                              hipStream_t stream) {
    const float* query = (const float*)d_in[0];   // [B,S,F]
    const float* value = (const float*)d_in[1];   // [B,S,F]  (key = value)
    const float* wq    = (const float*)d_in[2];   // [F, HQ*D]
    const float* wk    = (const float*)d_in[3];   // [F, HKV*D]
    const float* wv    = (const float*)d_in[4];   // [F, HKV*D]
    const float* wo    = (const float*)d_in[5];   // [HQ*D, F]
    float* out = (float*)d_out;                   // [B,S,F]

    // bf16 workspace layout (80 MB total)
    unsigned short* Qb = (unsigned short*)d_ws;                    // [B*S, HQ*D]
    unsigned short* Kc = Qb + (size_t)BB * SS * QKD;               // [B*S, HKV*D]
    unsigned short* Vc = Kc + (size_t)BB * SS * KVD;               // [B*S, HKV*D]
    unsigned short* Ab = Vc + (size_t)BB * SS * KVD;               // [B*S, HQ*D]

    const int M = BB * SS;                 // 4096
    const float qscale = 0.08838834764831845f;   // 1/sqrt(128)
    dim3 blk(256);

    // 1) projections (f32 in, bf16 out, WMMA bf16)
    gemm_wmma<false, true><<<dim3(QKD / 128, M / 128), blk, 0, stream>>>(
        query, FF, wq, QKD, Qb, QKD, FF, qscale);
    gemm_wmma<false, true><<<dim3(KVD / 128, M / 128), blk, 0, stream>>>(
        value, FF, wk, KVD, Kc, KVD, FF, 1.0f);
    gemm_wmma<false, true><<<dim3(KVD / 128, M / 128), blk, 0, stream>>>(
        value, FF, wv, KVD, Vc, KVD, FF, 1.0f);

    // 2) flash attention (bf16 in/out, async K staging)
    fa_kernel<<<dim3(SS / 128, HQ, BB), blk, 0, stream>>>(Qb, Kc, Vc, Ab);

    // 3) output projection (bf16 A, f32 out)
    gemm_wmma<true, false><<<dim3(FF / 128, M / 128), blk, 0, stream>>>(
        Ab, QKD, wo, FF, out, FF, QKD, 1.0f);
}
